// SelfAttention1D_82669530513868
// MI455X (gfx1250) — matrix operand order repaired and checked by reading, not compiled, not run on hardware
//
#include <hip/hip_runtime.h>
#include <hip/hip_bf16.h>

#define DID 1024
#define SEQ 2048
#define NH  16
#define HD  64
#define NB  2

typedef _Float16 __attribute__((ext_vector_type(4)))  v4h;
typedef _Float16 __attribute__((ext_vector_type(8)))  v8h;
typedef _Float16 __attribute__((ext_vector_type(16))) v16h;
typedef float    __attribute__((ext_vector_type(8)))  v8f;
typedef unsigned int __attribute__((ext_vector_type(4))) v4u;
typedef int      __attribute__((ext_vector_type(4)))  v4i;
typedef int      __attribute__((ext_vector_type(8)))  v8i;

__device__ __forceinline__ v16h cat16(v8h lo, v8h hi) {
  return __builtin_shufflevector(lo, hi, 0,1,2,3,4,5,6,7,8,9,10,11,12,13,14,15);
}

// A-fragment 16x32 f16 (MxK). lane<16: M=lane, K = k0+0..7 and k0+16..23
//                              lane>=16: M=lane-16, K = k0+8..15 and k0+24..31
__device__ __forceinline__ v16h load_a_frag(const _Float16* base, int row0, int ld,
                                            int k0, int lane) {
  int r  = row0 + (lane & 15);
  int kb = k0 + ((lane >> 4) << 3);
  const _Float16* p = base + (size_t)r * ld + kb;
  v8h lo = *(const v8h*)(p);
  v8h hi = *(const v8h*)(p + 16);
  return cat16(lo, hi);
}

// B-fragment 32x16 f16 (KxN), source is row-major M[n][k] (i.e. B = M^T).
// lane<16: N=lane, K = k0+0..7 and k0+16..23 ; lane>=16: N=lane-16, K = k0+8..15 and k0+24..31
__device__ __forceinline__ v16h load_b_frag(const _Float16* base, int n0, int ld,
                                            int k0, int lane) {
  int n  = n0 + (lane & 15);
  int kb = k0 + ((lane >> 4) << 3);
  const _Float16* p = base + (size_t)n * ld + kb;
  v8h lo = *(const v8h*)(p);
  v8h hi = *(const v8h*)(p + 16);
  return cat16(lo, hi);
}

__device__ __forceinline__ v8f wmma_f16(v16h a, v16h b, v8f c) {
  return __builtin_amdgcn_wmma_f32_16x16x32_f16(false, a, false, b, (short)0, c,
                                                false, false);
}

#define HAVE_TDM __has_builtin(__builtin_amdgcn_tensor_load_to_lds)

#if HAVE_TDM
// Issue a 2D TDM tile load (f16 elements): tile_d0 contiguous elems per row,
// tile_d1 rows with row stride `stride0` (elems), packed into LDS at lds_off.
// D# per CDNA5 ISA ch.8: group0 {count,lds,global,type=2}, group1 {dims/strides}.
// This toolchain exposes the 6-arg builtin:
//   (uint32x4 g0, int32x8 g1, int32x4 g2, int32x4 g3, int32x8 g4, i32 cpol)
__device__ __forceinline__ void tdm_load_2d(unsigned int lds_off, const void* gaddr,
                                            unsigned int tile_d0, unsigned int tile_d1,
                                            unsigned int stride0) {
  unsigned long long ga = (unsigned long long)gaddr;
  v4u g0;
  g0[0] = 1u;                                        // count=1, user descriptor
  g0[1] = lds_off;                                   // lds_addr (bytes)
  g0[2] = (unsigned int)(ga & 0xFFFFFFFFu);          // global_addr[31:0]
  g0[3] = (unsigned int)((ga >> 32) & 0x1FFFFFFu)    // global_addr[56:32]
          | (2u << 30);                              // type=2 ("image")
  const unsigned int td0 = 1u << 30, td1 = 1u << 30; // huge -> no OOB clipping
  v8i g1;
  g1[0] = (int)(1u << 16);                           // data_size = 2 bytes
  g1[1] = (int)((td0 & 0xFFFFu) << 16);              // tensor_dim0[15:0]
  g1[2] = (int)(((td0 >> 16) & 0xFFFFu) | ((td1 & 0xFFFFu) << 16));
  g1[3] = (int)(((td1 >> 16) & 0xFFFFu) | ((tile_d0 & 0xFFFFu) << 16));
  g1[4] = (int)(tile_d1 & 0xFFFFu);                  // tile_dim1, tile_dim2=0
  g1[5] = (int)stride0;                              // tensor_dim0_stride[31:0]
  g1[6] = 0;
  g1[7] = 0;
  v4i z4 = {0, 0, 0, 0};
  v8i z8 = {0, 0, 0, 0, 0, 0, 0, 0};
  __builtin_amdgcn_tensor_load_to_lds(g0, g1, z4, z4, z8, 0);
}
#else
// Fallback: cooperative synchronous staging (16B chunks across the block).
__device__ __forceinline__ void coop_load_tile(_Float16* dst, const _Float16* src,
                                               int rows, int rowlen, int src_ld,
                                               int tid, int nthreads) {
  const int cpr = rowlen >> 3;
  for (int c = tid; c < rows * cpr; c += nthreads) {
    int rr = c / cpr, cc = (c % cpr) << 3;
    *(v8h*)(dst + rr * rowlen + cc) = *(const v8h*)(src + (size_t)rr * src_ld + cc);
  }
}
#endif

// ---------------------------------------------------------------- f32 -> f16
__global__ __launch_bounds__(256) void cvt_f32_f16(const float* __restrict__ s,
                                                   _Float16* __restrict__ d, int n4) {
  int i = blockIdx.x * blockDim.x + threadIdx.x;
  if (i < n4) {
    float4 f = ((const float4*)s)[i];
    v4h h;
    h[0] = (_Float16)f.x; h[1] = (_Float16)f.y;
    h[2] = (_Float16)f.z; h[3] = (_Float16)f.w;
    ((v4h*)d)[i] = h;
  }
}

// ------------------------------------------------------- fused QKV projection
// Y = X @ W^T + b ; mode 0: Q (scaled 1/8, [BS,D] f16), 1: K ([BS,D] f16),
// 2: V stored transposed per batch as Vt[b][d][s] f16.
__global__ __launch_bounds__(128) void qkv_gemm(const _Float16* __restrict__ X,
                                                const _Float16* __restrict__ Wh,
                                                const float* __restrict__ bq,
                                                const float* __restrict__ bk,
                                                const float* __restrict__ bv,
                                                _Float16* __restrict__ Qh,
                                                _Float16* __restrict__ Kh,
                                                _Float16* __restrict__ Vt) {
  const int lane = threadIdx.x & 31;
  const int wave = threadIdx.x >> 5;
  const int mode = blockIdx.z;
  const int m0 = blockIdx.x * 128 + wave * 32;
  const int n0 = blockIdx.y * 64;
  const _Float16* W = Wh + (size_t)mode * DID * DID;
  const float* bias = (mode == 0) ? bq : (mode == 1 ? bk : bv);

  v8f acc[2][4] = {};
  for (int k0 = 0; k0 < DID; k0 += 32) {
    v16h a0 = load_a_frag(X, m0,      DID, k0, lane);
    v16h a1 = load_a_frag(X, m0 + 16, DID, k0, lane);
#pragma unroll
    for (int j = 0; j < 4; ++j) {
      v16h b = load_b_frag(W, n0 + 16 * j, DID, k0, lane);
      acc[0][j] = wmma_f16(a0, b, acc[0][j]);
      acc[1][j] = wmma_f16(a1, b, acc[1][j]);
    }
  }

  const int colx = lane & 15;
  const int rowb = (lane >> 4) << 3;
#pragma unroll
  for (int j = 0; j < 4; ++j) {
    const int col = n0 + 16 * j + colx;
    const float bb = bias[col];
#pragma unroll
    for (int i = 0; i < 2; ++i) {
      const int row = m0 + 16 * i + rowb;
      if (mode == 2) {
        const int b = row >> 11, s = row & (SEQ - 1);
        v8h pk;
#pragma unroll
        for (int e = 0; e < 8; ++e) pk[e] = (_Float16)(acc[i][j][e] + bb);
        *(v8h*)(Vt + ((size_t)b * DID + col) * SEQ + s) = pk;
      } else if (mode == 0) {
#pragma unroll
        for (int e = 0; e < 8; ++e)
          Qh[(size_t)(row + e) * DID + col] = (_Float16)((acc[i][j][e] + bb) * 0.125f);
      } else {
#pragma unroll
        for (int e = 0; e < 8; ++e)
          Kh[(size_t)(row + e) * DID + col] = (_Float16)(acc[i][j][e] + bb);
      }
    }
  }
}

// ------------------------------------------------------- flash attention
// grid: (SEQ/64, NH, NB), 128 threads = 4 waves; wave owns 16 query rows.
// K/V tiles are TDM-prefetched (double-buffered) into LDS and shared by all
// 4 waves; probabilities bounce through LDS to re-layout into A-fragments.
__global__ __launch_bounds__(128) void flash_attn(const _Float16* __restrict__ Qh,
                                                  const _Float16* __restrict__ Kh,
                                                  const _Float16* __restrict__ Vt,
                                                  const float* __restrict__ Bbias,
                                                  float* __restrict__ out) {
  __shared__ _Float16 ldsK[2][32 * 64];   // [buf][key-local][d-local]
  __shared__ _Float16 ldsV[2][64 * 32];   // [buf][d-local][key-local]
  __shared__ _Float16 ldsP[4][16 * 32];   // per-wave probability scratch
  const int lane = threadIdx.x & 31;
  const int wave = threadIdx.x >> 5;
  const int b = blockIdx.z, h = blockIdx.y;
  const int q0 = blockIdx.x * 64 + wave * 16;
  const int hoff = h * HD;
  const size_t batch_off = (size_t)b * SEQ * DID;
  const float Bh = Bbias[h];

  const _Float16* Qb = Qh + batch_off;
  const _Float16* Kb = Kh + batch_off;
  const _Float16* Vb = Vt + (size_t)b * DID * SEQ;

  v16h qa0 = load_a_frag(Qb, q0, DID, hoff,      lane);
  v16h qa1 = load_a_frag(Qb, q0, DID, hoff + 32, lane);

  v8f acc[4] = {};
  float mrow[8], lrow[8];
#pragma unroll
  for (int r = 0; r < 8; ++r) { mrow[r] = -__builtin_inff(); lrow[r] = 0.f; }

  const int colx = lane & 15;
  const int rowb = (lane >> 4) << 3;
  const int NSTEP = SEQ / 32;

#if HAVE_TDM
  if (wave == 0) {  // prologue: prefetch step 0 (TDM ignores EXEC; per-wave op)
    tdm_load_2d((unsigned int)(unsigned long long)&ldsK[0][0],
                Kb + /*key0=*/0 * DID + hoff, HD, 32, DID);
    tdm_load_2d((unsigned int)(unsigned long long)&ldsV[0][0],
                Vb + (size_t)hoff * SEQ + /*key0=*/0, 32, HD, SEQ);
  }
#endif

  for (int i = 0; i < NSTEP; ++i) {
    const int ib = i & 1;
    __syncthreads();   // A: prior reads of buf[ib^1] and ldsP are done
#if HAVE_TDM
    if (wave == 0) {
      if (i + 1 < NSTEP) {
        const int keyn = (i + 1) * 32;
        tdm_load_2d((unsigned int)(unsigned long long)&ldsK[ib ^ 1][0],
                    Kb + (size_t)keyn * DID + hoff, HD, 32, DID);
        tdm_load_2d((unsigned int)(unsigned long long)&ldsV[ib ^ 1][0],
                    Vb + (size_t)hoff * SEQ + keyn, 32, HD, SEQ);
        __builtin_amdgcn_s_wait_tensorcnt(2);  // step i's pair has landed
      } else {
        __builtin_amdgcn_s_wait_tensorcnt(0);
      }
    }
#else
    {
      const int key0 = i * 32;
      coop_load_tile(&ldsK[ib][0], Kb + (size_t)key0 * DID + hoff, 32, HD, DID,
                     threadIdx.x, 128);
      coop_load_tile(&ldsV[ib][0], Vb + (size_t)hoff * SEQ + key0, HD, 32, SEQ,
                     threadIdx.x, 128);
    }
#endif
    __syncthreads();   // B: buf[ib] ready for all waves

    // scores: 16q x 32k via two 16x16 tiles, K-dim = head_dim (two 32-steps)
    v8f s0 = {}, s1 = {};
    {
      v16h kb0 = load_b_frag(&ldsK[ib][0], 0,  HD, 0,  lane);
      v16h kb1 = load_b_frag(&ldsK[ib][0], 16, HD, 0,  lane);
      s0 = wmma_f16(qa0, kb0, s0);
      s1 = wmma_f16(qa0, kb1, s1);
      v16h kc0 = load_b_frag(&ldsK[ib][0], 0,  HD, 32, lane);
      v16h kc1 = load_b_frag(&ldsK[ib][0], 16, HD, 32, lane);
      s0 = wmma_f16(qa1, kc0, s0);
      s1 = wmma_f16(qa1, kc1, s1);
    }
    // online softmax (rows live across 16-lane groups; VGPR r = row rowb+r)
    float al[8];
#pragma unroll
    for (int r = 0; r < 8; ++r) {
      float x0 = s0[r] + Bh, x1 = s1[r] + Bh;
      float mx = fmaxf(x0, x1);
#pragma unroll
      for (int off = 1; off < 16; off <<= 1) mx = fmaxf(mx, __shfl_xor(mx, off, 16));
      float mnew = fmaxf(mrow[r], mx);
      float alpha = __expf(mrow[r] - mnew);
      float p0 = __expf(x0 - mnew);
      float p1 = __expf(x1 - mnew);
      float sum = p0 + p1;
#pragma unroll
      for (int off = 1; off < 16; off <<= 1) sum += __shfl_xor(sum, off, 16);
      lrow[r] = lrow[r] * alpha + sum;
      mrow[r] = mnew;
      al[r] = alpha;
      const int row = rowb + r;
      ldsP[wave][row * 32 + colx]      = (_Float16)p0;
      ldsP[wave][row * 32 + 16 + colx] = (_Float16)p1;
    }
#pragma unroll
    for (int j = 0; j < 4; ++j)
#pragma unroll
      for (int r = 0; r < 8; ++r) acc[j][r] *= al[r];

    __syncthreads();   // C: ldsP visible
    v16h pa = load_a_frag(&ldsP[wave][0], 0, 32, 0, lane);
#pragma unroll
    for (int j = 0; j < 4; ++j) {
      v16h vb = load_b_frag(&ldsV[ib][0], 16 * j, 32, 0, lane);
      acc[j] = wmma_f16(pa, vb, acc[j]);
    }
  }

#pragma unroll
  for (int j = 0; j < 4; ++j)
#pragma unroll
    for (int e = 0; e < 8; ++e) {
      const int srow = q0 + rowb + e;
      out[batch_off + (size_t)srow * DID + hoff + 16 * j + colx] = acc[j][e] / lrow[e];
    }
}

extern "C" void kernel_launch(void* const* d_in, const int* in_sizes, int n_in,
                              void* d_out, int out_size, void* d_ws, size_t ws_size,
                              hipStream_t stream) {
  const float* x  = (const float*)d_in[0];
  const float* Wq = (const float*)d_in[1];
  const float* bq = (const float*)d_in[2];
  const float* Wk = (const float*)d_in[3];
  const float* bk = (const float*)d_in[4];
  const float* Wv = (const float*)d_in[5];
  const float* bv = (const float*)d_in[6];
  const float* Bb = (const float*)d_in[7];
  float* out = (float*)d_out;

  const size_t NX = (size_t)NB * SEQ * DID;   // 4194304
  const size_t NW = (size_t)DID * DID;        // 1048576

  _Float16* xh = (_Float16*)d_ws;
  _Float16* Wh = xh + NX;          // 3 weights back-to-back
  _Float16* Qh = Wh + 3 * NW;
  _Float16* Kh = Qh + NX;
  _Float16* Vt = Kh + NX;

  cvt_f32_f16<<<(int)(NX / 4 / 256), 256, 0, stream>>>(x,  xh, (int)(NX / 4));
  cvt_f32_f16<<<(int)(NW / 4 / 256), 256, 0, stream>>>(Wq, Wh,          (int)(NW / 4));
  cvt_f32_f16<<<(int)(NW / 4 / 256), 256, 0, stream>>>(Wk, Wh + NW,     (int)(NW / 4));
  cvt_f32_f16<<<(int)(NW / 4 / 256), 256, 0, stream>>>(Wv, Wh + 2 * NW, (int)(NW / 4));

  dim3 ggrid(NB * SEQ / 128, DID / 64, 3);
  qkv_gemm<<<ggrid, 128, 0, stream>>>(xh, Wh, bq, bk, bv, Qh, Kh, Vt);

  dim3 agrid(SEQ / 64, NH, NB);
  flash_attn<<<agrid, 128, 0, stream>>>(Qh, Kh, Vt, Bb, out);
}
